// MOENet_56195352101425
// MI455X (gfx1250) — compile-verified, matrix operand order checked by reading, not run on hardware
//
#include <hip/hip_runtime.h>
#include <hip/hip_bf16.h>
#include <math.h>

// MoE conv net for MI455X (gfx1250, wave32).
// ws layout (floats): y[8*64*16384] | pmax[512] | psum[512] | cof[32]  (~33.6 MB)

#define DIM   64
#define CIN   128
#define NB    8
#define HH    128
#define WW    128
#define NPIX  (HH * WW)      // 16384
#define NEXP  4

typedef __bf16 bf16_t;
typedef __attribute__((ext_vector_type(16))) __bf16 v16bf;
typedef __attribute__((ext_vector_type(8)))  __bf16 v8bf;
typedef __attribute__((ext_vector_type(8)))  float  v8f;
typedef __attribute__((ext_vector_type(4)))  unsigned int v4u;
typedef __attribute__((ext_vector_type(8)))  int v8i;
typedef __attribute__((ext_vector_type(4)))  int v4i;

#define AS1 __attribute__((address_space(1)))
#define AS3 __attribute__((address_space(3)))

// ---- CDNA5 feature probes -------------------------------------------------
#if __has_builtin(__builtin_amdgcn_tensor_load_to_lds)
#define HAVE_TDM 1
#else
#define HAVE_TDM 0
#endif

#if __has_builtin(__builtin_amdgcn_global_load_async_to_lds_b32)
#define HAVE_ASYNC 1
#else
#define HAVE_ASYNC 0
#endif

// Builtin wants non-const AS(1)/AS(3) int* operands (per hipcc diagnostic).
static __device__ __forceinline__ AS1 int* gcast_i(const void* p) {
    return (AS1 int*)(unsigned long long)(uintptr_t)p;
}
static __device__ __forceinline__ AS3 int* lcast_i(void* p) {
    // generic LDS pointer: low 32 bits are the LDS byte offset (aperture rule)
    return (AS3 int*)(unsigned int)(uintptr_t)p;
}
static __device__ __forceinline__ unsigned int lds_off(const void* p) {
    return (unsigned int)(uintptr_t)p;
}

static __device__ __forceinline__ void wait_async0() {
#if __has_builtin(__builtin_amdgcn_s_wait_asynccnt)
    __builtin_amdgcn_s_wait_asynccnt(0);
#elif HAVE_ASYNC
    asm volatile("s_wait_asynccnt 0x0" ::: "memory");
#endif
}

#if HAVE_TDM
// Issue a 2D TDM tile load: tile_d0 x tile_d1 elements of 4 bytes,
// row stride (elements) = stride0, from `src` into LDS at `lds_dst`.
static __device__ __forceinline__ void tdm_load_2d_f32(
    const void* src, void* lds_dst,
    unsigned tensor_d0, unsigned tensor_d1,
    unsigned tile_d0, unsigned tile_d1, unsigned long long stride0)
{
    const unsigned long long ga = (unsigned long long)(uintptr_t)src;
    v4u g0;
    g0.x = 1u;                                       // count=1, user descriptor
    g0.y = lds_off(lds_dst);                         // lds_addr (bytes)
    g0.z = (unsigned)(ga & 0xffffffffu);             // global_addr[31:0]
    g0.w = (unsigned)((ga >> 32) & 0x01ffffffu)      // global_addr[56:32]
         | 0x80000000u;                              // type=2 ("image") in [127:126]
    v8i g1;
    g1[0] = (int)(2u << 16);                         // workgroup_mask=0, data_size=2 (4B)
    g1[1] = (int)((tensor_d0 & 0xffffu) << 16);      // tensor_dim0[15:0] -> bits[63:48]
    g1[2] = (int)(((tensor_d0 >> 16) & 0xffffu)      // tensor_dim0[31:16]
         | ((tensor_d1 & 0xffffu) << 16));           // tensor_dim1[15:0]
    g1[3] = (int)(((tensor_d1 >> 16) & 0xffffu)      // tensor_dim1[31:16]
         | ((tile_d0 & 0xffffu) << 16));             // tile_dim0 -> bits[127:112]
    g1[4] = (int)(tile_d1 & 0xffffu);                // tile_dim1; tile_dim2=0
    g1[5] = (int)(unsigned)(stride0 & 0xffffffffull);// tensor_dim0_stride[31:0]
    g1[6] = (int)(unsigned)((stride0 >> 32) & 0xffffull); // dim0_stride[47:32]
    g1[7] = 0;                                       // tensor_dim1_stride (unused, 2D)
    v4i zz = {0, 0, 0, 0};
#if __clang_major__ >= 23
    v8i zz8 = {0, 0, 0, 0, 0, 0, 0, 0};
    __builtin_amdgcn_tensor_load_to_lds(g0, g1, zz, zz, zz8, 0);
#else
    __builtin_amdgcn_tensor_load_to_lds(g0, g1, zz, zz, 0);
#endif
}
#endif

// ---------------------------------------------------------------------------
// Kernel 1: pre_fuse 1x1 conv as bf16 WMMA GEMM:
//   y[b,m,p] = sum_c W[m,c]*x[b,c,p] + bias[m]
// Block: 256 threads (8 waves); per block: 64 channels x 64 pixels.
// X slab staged by the Tensor Data Mover (TDM) when available.
// ---------------------------------------------------------------------------
__global__ __launch_bounds__(256)
void prefuse_wmma(const float* __restrict__ x, const float* __restrict__ pre_w,
                  const float* __restrict__ pre_b, float* __restrict__ y)
{
    __shared__ __align__(16) float  sXf[CIN * 64];   // f32 slab [c][p]
    __shared__ __align__(16) bf16_t sW [DIM * CIN];  // [m][k]
    __shared__ __align__(16) bf16_t sXT[64 * CIN];   // [p][k] (transposed bf16)

    const int t   = threadIdx.x;
    const int bid = blockIdx.x;
    const int b   = bid >> 8;                // 256 pixel-tiles per batch
    const int p0  = (bid & 255) << 6;        // 64 pixels per block

    const float* xb = x + (size_t)b * CIN * NPIX + p0;

#if HAVE_TDM
    // One DMA per block: 64x128 f32 tile, row stride 16384 elements.
    if (t < 32) {  // wave 0 issues; EXEC ignored by TDM, one issue per block
        tdm_load_2d_f32(xb, sXf, NPIX, CIN, 64, CIN, NPIX);
    }
#else
    __builtin_prefetch(xb, 0, 0);            // -> global_prefetch_b8
    for (int i = t; i < CIN * 64; i += 256) {
        int c = i >> 6, p = i & 63;
        sXf[c * 64 + p] = xb[(size_t)c * NPIX + p];
    }
#endif

    // Stage weights (f32 -> bf16) while TDM streams the slab.
    for (int i = t; i < DIM * CIN; i += 256)
        sW[i] = (bf16_t)pre_w[i];

#if HAVE_TDM
    __builtin_amdgcn_s_wait_tensorcnt(0);
#endif
    __syncthreads();

    // Transpose + convert: sXT[p][c] = bf16(sXf[c][p])
    for (int i = t; i < CIN * 64; i += 256) {
        int c = i >> 6, p = i & 63;
        sXT[p * CIN + c] = (bf16_t)sXf[c * 64 + p];
    }
    __syncthreads();

    const int w    = t >> 5;
    const int lane = t & 31;
    const int mt   = w & 3;          // M tile -> channels mt*16..+15
    const int nt   = w >> 2;         // N group (0..1)
    const int ln   = lane & 15;
    const int hi   = lane >> 4;
    const int m0   = mt * 16;
    const int n0a  = nt * 16;
    const int n0b  = nt * 16 + 32;

    v8f acc0 = {};
    v8f acc1 = {};

    #pragma unroll
    for (int ks = 0; ks < 4; ++ks) {
        const int kb = ks * 32;

        const v8bf* ap = (const v8bf*)&sW[(m0 + ln) * CIN + kb + hi * 8];
        v8bf a_lo = ap[0];
        v8bf a_hi = ap[2];
        v16bf a;
        #pragma unroll
        for (int e = 0; e < 8; ++e) { a[e] = a_lo[e]; a[e + 8] = a_hi[e]; }

        const v8bf* bpA = (const v8bf*)&sXT[(n0a + ln) * CIN + kb + hi * 8];
        v8bf bA_lo = bpA[0];
        v8bf bA_hi = bpA[2];
        v16bf bA;
        #pragma unroll
        for (int e = 0; e < 8; ++e) { bA[e] = bA_lo[e]; bA[e + 8] = bA_hi[e]; }

        acc0 = __builtin_amdgcn_wmma_f32_16x16x32_bf16(
                   false, a, false, bA, (short)0, acc0, false, false);

        const v8bf* bpB = (const v8bf*)&sXT[(n0b + ln) * CIN + kb + hi * 8];
        v8bf bB_lo = bpB[0];
        v8bf bB_hi = bpB[2];
        v16bf bB;
        #pragma unroll
        for (int e = 0; e < 8; ++e) { bB[e] = bB_lo[e]; bB[e + 8] = bB_hi[e]; }

        acc1 = __builtin_amdgcn_wmma_f32_16x16x32_bf16(
                   false, a, false, bB, (short)0, acc1, false, false);
    }

    float* yb = y + (size_t)(b * DIM) * NPIX + p0;
    #pragma unroll
    for (int r = 0; r < 8; ++r) {
        const int m = m0 + r + hi * 8;
        const float bias = pre_b[m];
        yb[(size_t)m * NPIX + n0a + ln] = acc0[r] + bias;
        yb[(size_t)m * NPIX + n0b + ln] = acc1[r] + bias;
    }
}

// ---------------------------------------------------------------------------
// Kernel 2: per-(b,c) max and sum over H*W
// ---------------------------------------------------------------------------
__global__ __launch_bounds__(256)
void pool_reduce(const float* __restrict__ y, float* __restrict__ pmax,
                 float* __restrict__ psum)
{
    __shared__ float smx[256];
    __shared__ float ssm[256];
    const int t   = threadIdx.x;
    const int bid = blockIdx.x;           // b*64 + c
    const float* p = y + (size_t)bid * NPIX;

    float mx = -INFINITY, sm = 0.f;
    for (int i = t; i < NPIX; i += 256) {
        float v = p[i];
        mx = fmaxf(mx, v);
        sm += v;
    }
    smx[t] = mx; ssm[t] = sm;
    __syncthreads();
    for (int s = 128; s > 0; s >>= 1) {
        if (t < s) {
            smx[t] = fmaxf(smx[t], smx[t + s]);
            ssm[t] += ssm[t + s];
        }
        __syncthreads();
    }
    if (t == 0) { pmax[bid] = smx[0]; psum[bid] = ssm[0]; }
}

// ---------------------------------------------------------------------------
// Kernel 3: gate network. One wave; lane b handles batch b.
// ---------------------------------------------------------------------------
__global__ __launch_bounds__(32)
void gate_kernel(const float* __restrict__ pmax, const float* __restrict__ psum,
                 const float* __restrict__ gw0, const float* __restrict__ gb0,
                 const float* __restrict__ gw1, const float* __restrict__ gb1,
                 float* __restrict__ cof)
{
    const int b = threadIdx.x;
    if (b >= NB) return;

    float logits[NEXP], noise[NEXP];
    #pragma unroll
    for (int e = 0; e < NEXP; ++e) {
        float la = gb1[e], na = gb0[e];
        for (int c = 0; c < DIM; ++c) {
            float pl = pmax[b * DIM + c] + psum[b * DIM + c] * (1.f / NPIX);
            la = fmaf(pl, gw1[e * DIM + c], la);
            na = fmaf(pl, gw0[e * DIM + c], na);
        }
        logits[e] = la > 0.f ? la : 0.2f * la;               // leaky_relu(0.2)
        noise[e]  = (na > 20.f) ? na : log1pf(expf(na));     // softplus
    }

    float mean = 0.f;
    #pragma unroll
    for (int e = 0; e < NEXP; ++e) mean += noise[e];
    mean *= 0.25f;
    float var = 0.f;
    #pragma unroll
    for (int e = 0; e < NEXP; ++e) { float d = noise[e] - mean; var = fmaf(d, d, var); }
    const float sd = sqrtf(var * (1.f / 3.f));               // ddof=1

    float scores[NEXP];
    #pragma unroll
    for (int e = 0; e < NEXP; ++e) scores[e] = logits[e] + (noise[e] - mean) / sd;

    int i0 = 0;
    #pragma unroll
    for (int e = 1; e < NEXP; ++e) if (scores[e] > scores[i0]) i0 = e;
    int i1 = -1;
    #pragma unroll
    for (int e = 0; e < NEXP; ++e)
        if (e != i0 && (i1 < 0 || scores[e] > scores[i1])) i1 = e;

    const float l0 = (i0 == 0) ? logits[0] : (i0 == 1) ? logits[1] : (i0 == 2) ? logits[2] : logits[3];
    const float l1 = (i1 == 0) ? logits[0] : (i1 == 1) ? logits[1] : (i1 == 2) ? logits[2] : logits[3];
    const float mx = fmaxf(l0, l1);
    const float e0 = expf(l0 - mx), e1 = expf(l1 - mx);
    const float inv = 1.f / (e0 + e1);

    #pragma unroll
    for (int e = 0; e < NEXP; ++e) cof[b * NEXP + e] = 0.f;
    cof[b * NEXP + i0] = e0 * inv;
    cof[b * NEXP + i1] = e1 * inv;
}

// ---------------------------------------------------------------------------
// Kernel 4: fused experts: dw3x3 -> exact GELU -> dw3x3, top-2 experts only.
// 32x32 output tile per block, 36x36 halo staged via async-to-LDS.
// ---------------------------------------------------------------------------
__global__ __launch_bounds__(256)
void experts_fused(const float* __restrict__ y, const float* __restrict__ cof,
                   const float* __restrict__ ew1, const float* __restrict__ eb1,
                   const float* __restrict__ ew2, const float* __restrict__ eb2,
                   float* __restrict__ out)
{
    __shared__ float sIn[36 * 36];
    __shared__ float sMid[34 * 34];

    const int t = threadIdx.x;
    int bid = blockIdx.x;
    const int tile = bid & 15;  bid >>= 4;
    const int c    = bid & 63;
    const int b    = bid >> 6;
    const int oy0  = (tile >> 2) * 32;
    const int ox0  = (tile & 3) * 32;

    const float* yp = y + (size_t)(b * DIM + c) * NPIX;

    // Stage 36x36 halo tile: valid lanes stream via async-to-LDS (EXEC-masked),
    // out-of-image lanes write the zero pad directly.
    for (int i = t; i < 36 * 36; i += 256) {
        const int iy = i / 36, ix = i - iy * 36;
        const int gy = oy0 + iy - 2, gx = ox0 + ix - 2;
        if ((unsigned)gy < (unsigned)HH && (unsigned)gx < (unsigned)WW) {
#if HAVE_ASYNC
            __builtin_amdgcn_global_load_async_to_lds_b32(
                gcast_i(&yp[gy * WW + gx]), lcast_i(&sIn[i]), 0, 0);
#else
            sIn[i] = yp[gy * WW + gx];
#endif
        } else {
            sIn[i] = 0.f;
        }
    }
    wait_async0();

    float acc[4] = {0.f, 0.f, 0.f, 0.f};

    for (int e = 0; e < NEXP; ++e) {
        const float cw = cof[b * NEXP + e];     // uniform across block
        if (cw == 0.f) continue;                // skip non-selected experts

        const float* w1 = ew1 + (size_t)(e * DIM + c) * 9;
        const float* w2 = ew2 + (size_t)(e * DIM + c) * 9;
        const float b1 = eb1[e * DIM + c];
        const float b2 = eb2[e * DIM + c];
        float k1[9], k2[9];
        #pragma unroll
        for (int k = 0; k < 9; ++k) { k1[k] = w1[k]; k2[k] = w2[k]; }

        __syncthreads();   // sIn ready / previous sMid consumers done

        // mid(34x34) = gelu(conv1(sIn)); positions outside image -> 0 (SAME pad)
        for (int i = t; i < 34 * 34; i += 256) {
            const int my = i / 34, mx = i - my * 34;
            const int gy = oy0 + my - 1, gx = ox0 + mx - 1;
            float g = 0.f;
            if ((unsigned)gy < (unsigned)HH && (unsigned)gx < (unsigned)WW) {
                float s = b1;
                #pragma unroll
                for (int ky = 0; ky < 3; ++ky)
                    #pragma unroll
                    for (int kx = 0; kx < 3; ++kx)
                        s = fmaf(k1[ky * 3 + kx], sIn[(my + ky) * 36 + (mx + kx)], s);
                g = 0.5f * s * (1.f + erff(s * 0.70710678118654752f));  // exact GELU
            }
            sMid[i] = g;
        }
        __syncthreads();

        // out(32x32) += cw * (conv2(mid) + b2)
        #pragma unroll
        for (int j = 0; j < 4; ++j) {
            const int idx = t + j * 256;
            const int oy = idx >> 5, ox = idx & 31;
            float s = b2;
            #pragma unroll
            for (int ky = 0; ky < 3; ++ky)
                #pragma unroll
                for (int kx = 0; kx < 3; ++kx)
                    s = fmaf(k2[ky * 3 + kx], sMid[(oy + ky) * 34 + (ox + kx)], s);
            acc[j] = fmaf(cw, s, acc[j]);
        }
    }

    float* op = out + (size_t)(b * DIM + c) * NPIX;
    #pragma unroll
    for (int j = 0; j < 4; ++j) {
        const int idx = t + j * 256;
        const int oy = idx >> 5, ox = idx & 31;
        op[(oy0 + oy) * WW + (ox0 + ox)] = acc[j];
    }
}

// ---------------------------------------------------------------------------
extern "C" void kernel_launch(void* const* d_in, const int* in_sizes, int n_in,
                              void* d_out, int out_size, void* d_ws, size_t ws_size,
                              hipStream_t stream)
{
    const float* x     = (const float*)d_in[0];
    const float* pre_w = (const float*)d_in[1];
    const float* pre_b = (const float*)d_in[2];
    const float* gw0   = (const float*)d_in[3];
    const float* gb0   = (const float*)d_in[4];
    const float* gw1   = (const float*)d_in[5];
    const float* gb1   = (const float*)d_in[6];
    const float* ew1   = (const float*)d_in[7];
    const float* eb1   = (const float*)d_in[8];
    const float* ew2   = (const float*)d_in[9];
    const float* eb2   = (const float*)d_in[10];
    float* out = (float*)d_out;

    float* y    = (float*)d_ws;                       // 512 * 16384 floats
    float* pmax = y + (size_t)NB * DIM * NPIX;        // 512
    float* psum = pmax + NB * DIM;                    // 512
    float* cof  = psum + NB * DIM;                    // 32

    hipLaunchKernelGGL(prefuse_wmma, dim3(NB * (NPIX / 64)), dim3(256), 0, stream,
                       x, pre_w, pre_b, y);
    hipLaunchKernelGGL(pool_reduce, dim3(NB * DIM), dim3(256), 0, stream,
                       y, pmax, psum);
    hipLaunchKernelGGL(gate_kernel, dim3(1), dim3(32), 0, stream,
                       pmax, psum, gw0, gb0, gw1, gb1, cof);
    hipLaunchKernelGGL(experts_fused, dim3(NB * DIM * 16), dim3(256), 0, stream,
                       y, cof, ew1, eb1, ew2, eb2, out);
}
